// RecurrentLearningModel_6047313953299
// MI455X (gfx1250) — matrix operand order repaired and checked by reading, not compiled, not run on hardware
//
#include <hip/hip_runtime.h>
#include <hip/hip_bf16.h>

// ---------------- problem constants ----------------
#define N_CL 40000
#define F_IN 20
#define D_H  128
#define S_ST 512
#define NEGV (-1e30f)
#define NBLK 100          // blocks for the per-step matvec/reduction kernel

typedef __attribute__((ext_vector_type(16))) __bf16 v16bf;
typedef __attribute__((ext_vector_type(8)))  float  v8f;

// f32 -> bf16 with round-to-nearest-even (bit-level, always compiles)
static __device__ inline __bf16 f2bf(float f) {
  union { float f; unsigned u; } v; v.f = f;
  unsigned r = v.u + 0x7fffu + ((v.u >> 16) & 1u);
  unsigned short hs = (unsigned short)(r >> 16);
  __bf16 out; __builtin_memcpy(&out, &hs, 2); return out;
}

// ---- WMMA fragment loaders (wave32), contiguous K order k=16h+s shared by A and B ----
// A: 16x32 bf16 (MxK), row-major source, leading dim ldk.
// lanes 0-15: M=lane, elems 0..15 -> K=0..15
// lanes 16-31: M=lane-16, elems 0..15 -> K=16..31
static __device__ inline v16bf load_A_bf(const __bf16* src, int ldk, int lane) {
  int m  = lane & 15;
  int kb = (lane < 16) ? 0 : 16;
  const __bf16* row = src + m * ldk;
  v16bf a;
#pragma unroll
  for (int e = 0; e < 8; e++) a[e] = row[kb + e];
#pragma unroll
  for (int e = 0; e < 8; e++) a[8 + e] = row[kb + 8 + e];
  return a;
}

// B: 32x16 bf16 (KxN), row-major source, leading dim ldn.
// lanes 0-15: N=lane, elems 0..15 -> K=0..15 ; lanes 16-31: N=lane-16, K=16..31
static __device__ inline v16bf load_B_bf(const __bf16* src, int ldn, int lane) {
  int n  = lane & 15;
  int kb = (lane < 16) ? 0 : 16;
  v16bf b;
#pragma unroll
  for (int e = 0; e < 16; e++) b[e] = src[(kb + e) * ldn + n];
  return b;
}

// ---------------- weight prep: f32 -> bf16 (W1 zero-padded K 20->32) ----------------
__global__ void prep_weights(const float* __restrict__ W1, const float* __restrict__ W2,
                             __bf16* __restrict__ W1bf, __bf16* __restrict__ W2bf) {
  int i = blockIdx.x * 256 + threadIdx.x;            // 64*256 == 128*128
  if (i < 32 * D_H) {
    int r = i >> 7;                                  // row of padded [32,128]
    W1bf[i] = f2bf(r < F_IN ? W1[i] : 0.f);
  }
  W2bf[i] = f2bf(W2[i]);
}

__global__ void init_state_k(const float* __restrict__ ik, const float* __restrict__ is_,
                             float* __restrict__ h, float* __restrict__ c,
                             float* __restrict__ scal) {
  int i = threadIdx.x;
  h[i] = ik[i]; c[i] = is_[i];
  if (i < 2) scal[i] = 0.f;                          // scal[0]=cnt, scal[1]=sum(terms)
}

// ---------------- fused embedder: emb = relu(F@W1+b1)@W2+b2, WMMA bf16 ----------------
// One block = 16 rows of emb. 8 waves, wave w owns output columns [16w,16w+16).
__global__ __launch_bounds__(256) void embed_kernel(
    const float* __restrict__ feat, const __bf16* __restrict__ W1bf,
    const float* __restrict__ b1, const __bf16* __restrict__ W2bf,
    const float* __restrict__ b2, float* __restrict__ emb)
{
  __shared__ __bf16 Af[16 * 32];     // features tile, K padded to 32
  __shared__ __bf16 H1[16 * D_H];    // relu(F@W1+b1) tile in bf16

  const int tid  = threadIdx.x;
  const int wave = tid >> 5, lane = tid & 31;
  const int row0 = blockIdx.x * 16;  // N_CL % 16 == 0

  for (int i = tid; i < 16 * 32; i += 256) {
    int r = i >> 5, k = i & 31;
    float v = (k < F_IN) ? feat[(size_t)(row0 + r) * F_IN + k] : 0.f;
    Af[i] = f2bf(v);
  }
  __syncthreads();

  const int n    = lane & 15;
  const int moff = (lane < 16) ? 0 : 8;
  const int col  = 16 * wave + n;

  // stage 1: one K=32 WMMA per 16x16 tile (K padded 20->32 with zeros)
  {
    v16bf a = load_A_bf(Af, 32, lane);
    v16bf b = load_B_bf(W1bf + 16 * wave, D_H, lane);
    v8f acc = {};
    acc = __builtin_amdgcn_wmma_f32_16x16x32_bf16(false, a, false, b,
                                                  (short)0, acc, false, false);
    float bb = b1[col];
#pragma unroll
    for (int r = 0; r < 8; r++)
      H1[(moff + r) * D_H + col] = f2bf(fmaxf(acc[r] + bb, 0.f));
  }
  __syncthreads();

  // stage 2: K=128 -> 4 WMMA accumulating into f32
  {
    v8f acc = {};
#pragma unroll
    for (int kt = 0; kt < 4; kt++) {
      v16bf a = load_A_bf(H1 + kt * 32, D_H, lane);
      v16bf b = load_B_bf(W2bf + (size_t)kt * 32 * D_H + 16 * wave, D_H, lane);
      acc = __builtin_amdgcn_wmma_f32_16x16x32_bf16(false, a, false, b,
                                                    (short)0, acc, false, false);
    }
    float bb = b2[col];
#pragma unroll
    for (int r = 0; r < 8; r++)
      emb[(size_t)(row0 + moff + r) * D_H + col] = acc[r] + bb;
  }
}

// ---------------- per-step: masked logits + online-logsumexp block partials ----------------
// partials[b*8 + {0..5}] = {M, S=sum e^(x-M), T=sum e^(x-M)*x, n_pass, n_good, sum_good x}
__global__ __launch_bounds__(256) void scan_partials(
    const float* __restrict__ emb, const unsigned char* __restrict__ mask,
    const unsigned char* __restrict__ good, const float* __restrict__ h,
    float* __restrict__ partials)
{
  __shared__ float hs[D_H];
  __shared__ float rM[256], rS[256], rT[256], rNP[256], rNG[256], rCE[256];
  const int tid = threadIdx.x;
  if (tid < D_H) hs[tid] = h[tid];
  __syncthreads();

  float M = NEGV, S = 0.f, T = 0.f, np = 0.f, ng = 0.f, ce = 0.f;
  for (int row = blockIdx.x * 256 + tid; row < N_CL; row += 256 * NBLK) {
    if (!mask[row]) continue;                        // unmasked rows contribute 0 (e^(NEG-lse)=0)
    const float4* er = (const float4*)(emb + (size_t)row * D_H);
    float dot = 0.f;
#pragma unroll 8
    for (int k = 0; k < 32; k++) {
      float4 a = er[k];
      dot += a.x * hs[4*k] + a.y * hs[4*k+1] + a.z * hs[4*k+2] + a.w * hs[4*k+3];
    }
    np += 1.f;
    if (good[row]) { ng += 1.f; ce += dot; }
    if (dot > M) {
      float e = __expf(M - dot);
      S = S * e + 1.f;  T = T * e + dot;  M = dot;
    } else {
      float e = __expf(dot - M);
      S += e;  T += e * dot;
    }
  }
  rM[tid]=M; rS[tid]=S; rT[tid]=T; rNP[tid]=np; rNG[tid]=ng; rCE[tid]=ce;
  __syncthreads();
  for (int off = 128; off > 0; off >>= 1) {
    if (tid < off) {
      float M1=rM[tid], M2=rM[tid+off];
      float Mx = fmaxf(M1, M2);
      float a1 = __expf(M1 - Mx), a2 = __expf(M2 - Mx);
      rM[tid] = Mx;
      rS[tid] = rS[tid]*a1 + rS[tid+off]*a2;
      rT[tid] = rT[tid]*a1 + rT[tid+off]*a2;
      rNP[tid] += rNP[tid+off]; rNG[tid] += rNG[tid+off]; rCE[tid] += rCE[tid+off];
    }
    __syncthreads();
  }
  if (tid == 0) {
    float* p = partials + blockIdx.x * 8;
    p[0]=rM[0]; p[1]=rS[0]; p[2]=rT[0]; p[3]=rNP[0]; p[4]=rNG[0]; p[5]=rCE[0];
  }
}

// ---------------- per-step finalize: loss term + LSTM cell update ----------------
__global__ __launch_bounds__(512) void step_update(
    const float* __restrict__ emb, const int* __restrict__ sel_idx,
    const float* __restrict__ Wih, const float* __restrict__ Whh,
    const float* __restrict__ bih, const float* __restrict__ bhh,
    float* __restrict__ h, float* __restrict__ c,
    const float* __restrict__ partials, float* __restrict__ scal, int s)
{
  __shared__ float xs[D_H], hsv[D_H], gbuf[4 * D_H];
  const int tid = threadIdx.x;
  const int idx = sel_idx[s];
  if (tid < D_H) {
    hsv[tid] = h[tid];
    xs[tid]  = emb[(size_t)idx * D_H + tid];
  }
  if (tid == 0) {
    float M = NEGV, S = 0.f, T = 0.f, np = 0.f, ng = 0.f, ceS = 0.f;
    for (int b = 0; b < NBLK; b++) {
      const float* p = partials + b * 8;
      float Mx = fmaxf(M, p[0]);
      float a1 = __expf(M - Mx), a2 = __expf(p[0] - Mx);
      S = S*a1 + p[1]*a2;  T = T*a1 + p[2]*a2;  M = Mx;
      np += p[3]; ng += p[4]; ceS += p[5];
    }
    float term = 0.f;
    float nb = np - ng;
    float cnt = scal[0];
    if (ng > 0.f && nb > 0.f) {                       // "active" step
      float lse  = M + __logf(S);
      float cev  = -(ceS - ng * lse) / fmaxf(ng, 1.f);
      float ment = ((T - lse * S) / S) / __logf(fmaxf(np, 2.f));
      term = __powf(0.995f, cnt) * (nb / np) * cev + 0.1f * ment;
      scal[0] = cnt + 1.f;
    }
    scal[1] += term;
  }
  __syncthreads();
  // gates = Wih@x + bih + Whh@h + bhh ; thread j owns gate row j (512 rows, 256 MACs each)
  float g = bih[tid] + bhh[tid];
  const float* wi = Wih + (size_t)tid * D_H;
  const float* wh = Whh + (size_t)tid * D_H;
#pragma unroll 4
  for (int k = 0; k < D_H; k++) g += wi[k] * xs[k] + wh[k] * hsv[k];
  gbuf[tid] = g;
  __syncthreads();
  if (tid < D_H) {
    float gi = gbuf[tid], gf = gbuf[D_H + tid], gg = gbuf[2*D_H + tid], go = gbuf[3*D_H + tid];
    float si = 1.f / (1.f + __expf(-gi));
    float sf = 1.f / (1.f + __expf(-gf));
    float so = 1.f / (1.f + __expf(-go));
    float c2 = sf * c[tid] + si * tanhf(gg);
    h[tid] = so * tanhf(c2);
    c[tid] = c2;
  }
}

__global__ void loss_kernel(float* __restrict__ out, const float* __restrict__ scal) {
  out[0] = scal[1] / fmaxf(scal[0], 1.f);
}

// ---------------- launch ----------------
extern "C" void kernel_launch(void* const* d_in, const int* in_sizes, int n_in,
                              void* d_out, int out_size, void* d_ws, size_t ws_size,
                              hipStream_t stream) {
  const float*         feat   = (const float*)d_in[0];
  const unsigned char* selm   = (const unsigned char*)d_in[1];   // jax bool -> 1 byte
  const unsigned char* goodm  = (const unsigned char*)d_in[2];
  const int*           selidx = (const int*)d_in[3];
  const float* W1  = (const float*)d_in[4];
  const float* b1  = (const float*)d_in[5];
  const float* W2  = (const float*)d_in[6];
  const float* b2  = (const float*)d_in[7];
  const float* ik  = (const float*)d_in[8];
  const float* is_ = (const float*)d_in[9];
  const float* Wih = (const float*)d_in[10];
  const float* Whh = (const float*)d_in[11];
  const float* bih = (const float*)d_in[12];
  const float* bhh = (const float*)d_in[13];

  // workspace layout (all offsets 256B-aligned)
  char*   ws       = (char*)d_ws;
  float*  emb      = (float*)(ws);                       // 40000*128*4 = 20,480,000
  __bf16* W1bf     = (__bf16*)(ws + 20480000);           // 32*128*2   = 8,192
  __bf16* W2bf     = (__bf16*)(ws + 20488192);           // 128*128*2  = 32,768
  float*  h        = (float*)(ws + 20520960);            // 512
  float*  c        = (float*)(ws + 20521472);            // 512
  float*  partials = (float*)(ws + 20521984);            // NBLK*8*4 = 3,200
  float*  scal     = (float*)(ws + 20525312);            // cnt, term_sum

  prep_weights<<<64, 256, 0, stream>>>(W1, W2, W1bf, W2bf);
  init_state_k<<<1, 128, 0, stream>>>(ik, is_, h, c, scal);
  embed_kernel<<<N_CL / 16, 256, 0, stream>>>(feat, W1bf, b1, W2bf, b2, emb);

  for (int s = 0; s < S_ST; s++) {
    scan_partials<<<NBLK, 256, 0, stream>>>(emb, selm + (size_t)s * N_CL,
                                            goodm + (size_t)s * N_CL, h, partials);
    step_update<<<1, 512, 0, stream>>>(emb, selidx, Wih, Whh, bih, bhh,
                                       h, c, partials, scal, s);
  }
  loss_kernel<<<1, 1, 0, stream>>>((float*)d_out, scal);
}
